// ULSTM_Layer_48696339202288
// MI455X (gfx1250) — compile-verified, matrix operand order checked by reading, not run on hardware
//
#include <hip/hip_runtime.h>
#include <hip/hip_bf16.h>

// ---------------------------------------------------------------------------
// ULSTM layer for MI455X (gfx1250): bf16 WMMA (v_wmma_f32_16x16x32_bf16),
// f32 accumulation, f32 cell state.
//   Phase 0: pack weights + X into WMMA-fragment-native bf16 layouts.
//   Phase 1: one big GEMM  ux = X @ W_iozfux   (register-pipelined K loop)
//   Phase 2: 1024 sequential steps, 2 kernels each; the shared A operand
//            (h / z*tanh(c)) is staged per-block into LDS with
//            global_load_async_to_lds_b128 (ASYNCcnt) and read as ds_load.
// ---------------------------------------------------------------------------

#define SEQ   1024
#define BATCH 32
#define INDIM 1024
#define HDIM  1024
#define G4    (4 * HDIM)   // 4096
#define G5    (5 * HDIM)   // 5120
#define KT32  (HDIM / 32)  // 32 k-steps of 32

typedef __attribute__((ext_vector_type(16))) __bf16 v16bf;
typedef __attribute__((ext_vector_type(8)))  __bf16 v8bf;
typedef __attribute__((ext_vector_type(8)))  float  v8f;

union Frag16 { v16bf v; v8bf h[2]; };

__device__ __forceinline__ v8f wmma_bf16(v16bf a, v16bf b, v8f c) {
  // (neg_a, A, neg_b, B, c_mod, C, reuse_a, reuse_b)
  return __builtin_amdgcn_wmma_f32_16x16x32_bf16(false, a, false, b,
                                                 (short)0, c, false, false);
}

// CDNA5 async copy: 16 bytes global -> LDS, tracked by ASYNCcnt.
// LDS address operand = low 32 bits of the generic LDS pointer (ISA: LDS_ADDR
// = addr[31:0] after aperture check).
__device__ __forceinline__ void async_copy_b128(void* lds_dst, const void* gsrc) {
  asm volatile("global_load_async_to_lds_b128 %0, %1, off"
               :
               : "v"((unsigned)(unsigned long long)lds_dst), "v"(gsrc)
               : "memory");
}
__device__ __forceinline__ void wait_async0() {
  asm volatile("s_wait_asynccnt 0" ::: "memory");
}

// Load one 16x32 bf16 A-fragment from a row-major bf16 matrix (ld elems/row).
// ISA 16-bit A layout: lane<16 -> rows M=lane, K chunks [0,8)+[16,24);
//                      lane>=16 -> rows M=lane-16, K chunks [8,16)+[24,32).
// Works for both global and LDS (generic) pointers; from __shared__ arrays the
// compiler lowers the v8bf loads to ds_load_b128.
__device__ __forceinline__ v16bf load_a_rowmajor(const __bf16* base,
                                                 int ld, int kbase, int lane) {
  int m  = lane & 15;
  int b0 = (lane < 16) ? 0 : 8;
  const __bf16* p = base + (size_t)m * ld + kbase + b0;
  Frag16 f;
  f.h[0] = *(const v8bf*)(p);        // K = kbase+b0    .. +8
  f.h[1] = *(const v8bf*)(p + 16);   // K = kbase+b0+16 .. +24
  return f.v;
}

// Load a fragment from packed layout: [frag][lane][16 bf16] (32B/lane, contig).
__device__ __forceinline__ v16bf load_packed(const __bf16* __restrict__ base,
                                             int frag, int lane) {
  const __bf16* p = base + ((size_t)frag * 32 + lane) * 16;
  Frag16 f;
  f.h[0] = *(const v8bf*)(p);
  f.h[1] = *(const v8bf*)(p + 8);
  return f.v;
}

// ---------------- Phase 0: packing kernels ----------------

// Pack B operand: W (K=1024 rows, N cols, f32 row-major) -> [kt][nt][lane][16]
// fragment element e maps to K = kt*32 + b0 + (e<8 ? e : e+8), N = nt*16+lane%16.
__global__ __launch_bounds__(256) void pack_b_kernel(const float* __restrict__ W,
                                                     __bf16* __restrict__ out,
                                                     int N) {
  int tid  = blockIdx.x * blockDim.x + threadIdx.x;
  int lane = tid & 31;
  int frag = tid >> 5;
  int NT = N >> 4;
  if (frag >= (INDIM >> 5) * NT) return;
  int nt = frag % NT;
  int kt = frag / NT;
  int n  = nt * 16 + (lane & 15);
  int b0 = (lane < 16) ? 0 : 8;
  __bf16* dst = out + ((size_t)frag * 32 + lane) * 16;
#pragma unroll
  for (int e = 0; e < 16; ++e) {
    int K = kt * 32 + b0 + (e < 8 ? e : e + 8);
    dst[e] = (__bf16)W[(size_t)K * N + n];
  }
}

// Pack A operand: X (M=SEQ*BATCH rows, 1024 cols f32) -> [mt][kt][lane][16]
__global__ __launch_bounds__(256) void pack_a_kernel(const float* __restrict__ X,
                                                     __bf16* __restrict__ out) {
  int tid  = blockIdx.x * blockDim.x + threadIdx.x;
  int lane = tid & 31;
  int frag = tid >> 5;
  const int MT = (SEQ * BATCH) / 16;      // 2048
  if (frag >= MT * 32) return;
  int kt = frag & 31;
  int mt = frag >> 5;
  int m  = mt * 16 + (lane & 15);
  int b0 = (lane < 16) ? 0 : 8;
  __bf16* dst = out + ((size_t)frag * 32 + lane) * 16;
#pragma unroll
  for (int e = 0; e < 16; ++e) {
    int K = kt * 32 + b0 + (e < 8 ? e : e + 8);
    dst[e] = (__bf16)X[(size_t)m * INDIM + K];
  }
}

__global__ __launch_bounds__(256) void init_state_kernel(float* __restrict__ c,
                                                         float* __restrict__ ct,
                                                         __bf16* __restrict__ hbf) {
  int t = blockIdx.x * blockDim.x + threadIdx.x;
  if (t < BATCH * HDIM) {
    c[t]  = 0.0f;
    ct[t] = 0.0f;
    hbf[t] = (__bf16)0.0f;
  }
}

// ---------------- Phase 1: ux = X @ W_iozfux (bf16 out, bias added later) ---
// 16x64 tile per wave, K loop register double-buffered so VMEM latency hides
// under the WMMA issue stream (partial s_wait_loadcnt instead of wait-0).

__global__ __launch_bounds__(256) void gemm_ux_kernel(const __bf16* __restrict__ Xp,
                                                      const __bf16* __restrict__ Wp,
                                                      __bf16* __restrict__ ux) {
  int lane = threadIdx.x & 31;
  int wave = (blockIdx.x * blockDim.x + threadIdx.x) >> 5;
  const int NT64 = G5 / 64;               // 80
  int mt   = wave / NT64;                 // 0..2047
  int nt64 = wave % NT64;
  v8f acc0 = {}, acc1 = {}, acc2 = {}, acc3 = {};

  v16bf a_c  = load_packed(Xp, mt * 32, lane);
  v16bf b_c0 = load_packed(Wp, nt64 * 4 + 0, lane);
  v16bf b_c1 = load_packed(Wp, nt64 * 4 + 1, lane);
  v16bf b_c2 = load_packed(Wp, nt64 * 4 + 2, lane);
  v16bf b_c3 = load_packed(Wp, nt64 * 4 + 3, lane);
  for (int kt = 0; kt < KT32; ++kt) {
    int kn = (kt + 1 < KT32) ? kt + 1 : kt;   // clamped prefetch index
    v16bf a_n  = load_packed(Xp, mt * 32 + kn, lane);
    v16bf b_n0 = load_packed(Wp, kn * (G5 / 16) + nt64 * 4 + 0, lane);
    v16bf b_n1 = load_packed(Wp, kn * (G5 / 16) + nt64 * 4 + 1, lane);
    v16bf b_n2 = load_packed(Wp, kn * (G5 / 16) + nt64 * 4 + 2, lane);
    v16bf b_n3 = load_packed(Wp, kn * (G5 / 16) + nt64 * 4 + 3, lane);
    acc0 = wmma_bf16(a_c, b_c0, acc0);
    acc1 = wmma_bf16(a_c, b_c1, acc1);
    acc2 = wmma_bf16(a_c, b_c2, acc2);
    acc3 = wmma_bf16(a_c, b_c3, acc3);
    a_c = a_n; b_c0 = b_n0; b_c1 = b_n1; b_c2 = b_n2; b_c3 = b_n3;
  }
  // C/D layout: lane holds column n=lane%16, rows m = v + (lane>=16 ? 8 : 0)
  int n0 = nt64 * 64 + (lane & 15);
  int m0 = mt * 16 + ((lane < 16) ? 0 : 8);
  v8f accs[4] = {acc0, acc1, acc2, acc3};
#pragma unroll
  for (int j = 0; j < 4; ++j)
#pragma unroll
    for (int v = 0; v < 8; ++v)
      ux[(size_t)(m0 + v) * G5 + n0 + j * 16] = (__bf16)accs[j][v];
}

// ---------------- Phase 2a: gate GEMM + sigmoid -----------------------------
// G = sigmoid(ux_iozf + h @ W_iozfh + b_iozfux + b_iozfh)
// All 8 waves of a block share one 16-row h tile: stage it in LDS (32 KB)
// with async-to-LDS, then A fragments come from ds_load_b128.

__global__ __launch_bounds__(256) void lstm_gates_kernel(
    const __bf16* __restrict__ hbf, const __bf16* __restrict__ Whp,
    const __bf16* __restrict__ ux, const float* __restrict__ b_ux,
    const float* __restrict__ b_h, const float* __restrict__ ct,
    float* __restrict__ gates, __bf16* __restrict__ zct, int s) {
  __shared__ __bf16 hsh[16 * HDIM];                 // 32 KB
  int lane  = threadIdx.x & 31;
  int wavei = threadIdx.x >> 5;
  int mt = blockIdx.x >> 5;                          // uniform per block
  int nt = (blockIdx.x & 31) * 8 + wavei;            // 0..255

  const __bf16* hsrc = hbf + (size_t)mt * 16 * HDIM; // 16 contiguous rows
#pragma unroll
  for (int i = 0; i < 8; ++i) {
    int c16 = threadIdx.x + i * 256;                 // 2048 x 16B chunks
    async_copy_b128(hsh + c16 * 8, hsrc + c16 * 8);
  }
  wait_async0();
  __syncthreads();

  v8f acc = {};
  v16bf b_c = load_packed(Whp, nt, lane);
  for (int kt = 0; kt < KT32; ++kt) {
    int kn = (kt + 1 < KT32) ? kt + 1 : kt;
    v16bf b_n = load_packed(Whp, kn * 256 + nt, lane);
    v16bf a = load_a_rowmajor(hsh, HDIM, kt * 32, lane);   // ds_load path
    acc = wmma_bf16(a, b_c, acc);
    b_c = b_n;
  }

  int n  = nt * 16 + (lane & 15);
  int m0 = mt * 16 + ((lane < 16) ? 0 : 8);
  const __bf16* uxrow = ux + (size_t)s * BATCH * G5;
  float bias = b_ux[n] + b_h[n];
  bool is_z = (n >= 2 * HDIM) && (n < 3 * HDIM);     // uniform per wave
#pragma unroll
  for (int v = 0; v < 8; ++v) {
    int m = m0 + v;
    float pre = acc[v] + (float)uxrow[(size_t)m * G5 + n] + bias;
    float sg  = 1.0f / (1.0f + __expf(-pre));
    if (is_z) {
      int k = n - 2 * HDIM;
      zct[m * HDIM + k] = (__bf16)(sg * ct[m * HDIM + k]);
    } else {
      gates[m * G4 + n] = sg;
    }
  }
}

// ---------------- Phase 2b: u GEMM + cell/hidden update ---------------------
// u = tanh(ux_u + zct @ W_um + b); c' = i*u + f*c; ct' = tanh(c'); h = o*ct'

__global__ __launch_bounds__(256) void lstm_update_kernel(
    const __bf16* __restrict__ zct, const __bf16* __restrict__ Wump,
    const __bf16* __restrict__ ux, const float* __restrict__ b_ux,
    const float* __restrict__ b_um, const float* __restrict__ gates,
    float* __restrict__ c, float* __restrict__ ct, __bf16* __restrict__ hbf,
    float* __restrict__ out, int s) {
  __shared__ __bf16 zsh[16 * HDIM];                  // 32 KB
  int lane  = threadIdx.x & 31;
  int wavei = threadIdx.x >> 5;
  int mt = blockIdx.x >> 3;                          // uniform per block
  int nt = (blockIdx.x & 7) * 8 + wavei;             // 0..63

  const __bf16* zsrc = zct + (size_t)mt * 16 * HDIM;
#pragma unroll
  for (int i = 0; i < 8; ++i) {
    int c16 = threadIdx.x + i * 256;
    async_copy_b128(zsh + c16 * 8, zsrc + c16 * 8);
  }
  wait_async0();
  __syncthreads();

  v8f acc = {};
  v16bf b_c = load_packed(Wump, nt, lane);
  for (int kt = 0; kt < KT32; ++kt) {
    int kn = (kt + 1 < KT32) ? kt + 1 : kt;
    v16bf b_n = load_packed(Wump, kn * 64 + nt, lane);
    v16bf a = load_a_rowmajor(zsh, HDIM, kt * 32, lane);
    acc = wmma_bf16(a, b_c, acc);
    b_c = b_n;
  }

  int n  = nt * 16 + (lane & 15);
  int m0 = mt * 16 + ((lane < 16) ? 0 : 8);
  const __bf16* uxrow = ux + (size_t)s * BATCH * G5 + 4 * HDIM;
  float bias = b_ux[4 * HDIM + n] + b_um[n];
  float* outrow = out + (size_t)s * BATCH * HDIM;
#pragma unroll
  for (int v = 0; v < 8; ++v) {
    int m  = m0 + v;
    float u = tanhf(acc[v] + (float)uxrow[(size_t)m * G5 + n] + bias);
    float gi = gates[m * G4 + n];
    float go = gates[m * G4 + HDIM + n];
    float gf = gates[m * G4 + 3 * HDIM + n];
    float cn  = gi * u + gf * c[m * HDIM + n];
    float ctn = tanhf(cn);
    float hn  = go * ctn;
    c[m * HDIM + n]  = cn;
    ct[m * HDIM + n] = ctn;
    hbf[m * HDIM + n] = (__bf16)hn;
    outrow[(size_t)m * HDIM + n] = hn;
  }
}

// Tail: h_n = output[S-1], c_n = c
__global__ __launch_bounds__(256) void finish_kernel(const float* __restrict__ c,
                                                     float* __restrict__ out) {
  int t = blockIdx.x * blockDim.x + threadIdx.x;
  if (t < BATCH * HDIM) {
    out[(size_t)SEQ * BATCH * HDIM + t] = out[(size_t)(SEQ - 1) * BATCH * HDIM + t];
    out[(size_t)SEQ * BATCH * HDIM + BATCH * HDIM + t] = c[t];
  }
}

// ---------------------------------------------------------------------------

extern "C" void kernel_launch(void* const* d_in, const int* in_sizes, int n_in,
                              void* d_out, int out_size, void* d_ws, size_t ws_size,
                              hipStream_t stream) {
  (void)in_sizes; (void)n_in; (void)out_size; (void)ws_size;
  const float* X       = (const float*)d_in[0];
  const float* W_ux    = (const float*)d_in[1];
  const float* b_ux    = (const float*)d_in[2];
  const float* W_h     = (const float*)d_in[3];
  const float* b_h     = (const float*)d_in[4];
  const float* W_um    = (const float*)d_in[5];
  const float* b_um    = (const float*)d_in[6];
  float* out = (float*)d_out;

  char* ws = (char*)d_ws;
  size_t o = 0;
  auto carve = [&](size_t bytes) {
    size_t r = o;
    o = (o + bytes + 255) & ~(size_t)255;
    return r;
  };
  __bf16* uxbf = (__bf16*)(ws + carve((size_t)SEQ * BATCH * G5 * 2)); // 335 MB
  __bf16* Wuxp = (__bf16*)(ws + carve((size_t)INDIM * G5 * 2));       // 10 MB
  __bf16* Whp  = (__bf16*)(ws + carve((size_t)HDIM * G4 * 2));        // 8 MB
  __bf16* Wump = (__bf16*)(ws + carve((size_t)HDIM * HDIM * 2));      // 2 MB
  __bf16* Xp   = (__bf16*)(ws + carve((size_t)SEQ * BATCH * INDIM * 2)); // 64 MB
  __bf16* hbf  = (__bf16*)(ws + carve((size_t)BATCH * HDIM * 2));
  __bf16* zct  = (__bf16*)(ws + carve((size_t)BATCH * HDIM * 2));
  float*  gates= (float*) (ws + carve((size_t)BATCH * G4 * 4));
  float*  cbuf = (float*) (ws + carve((size_t)BATCH * HDIM * 4));
  float*  ctbuf= (float*) (ws + carve((size_t)BATCH * HDIM * 4));

  // Phase 0: pack (per call) + state init
  pack_b_kernel<<<(32 * (G5 / 16) * 32 + 255) / 256, 256, 0, stream>>>(W_ux, Wuxp, G5);
  pack_b_kernel<<<(32 * (G4 / 16) * 32 + 255) / 256, 256, 0, stream>>>(W_h, Whp, G4);
  pack_b_kernel<<<(32 * (HDIM / 16) * 32 + 255) / 256, 256, 0, stream>>>(W_um, Wump, HDIM);
  pack_a_kernel<<<(((SEQ * BATCH) / 16) * 32 * 32 + 255) / 256, 256, 0, stream>>>(X, Xp);
  init_state_kernel<<<(BATCH * HDIM + 255) / 256, 256, 0, stream>>>(cbuf, ctbuf, hbf);

  // Phase 1: input projection (2048 mtiles x 80 n64-tiles waves)
  {
    int waves = ((SEQ * BATCH) / 16) * (G5 / 64);
    gemm_ux_kernel<<<waves * 32 / 256, 256, 0, stream>>>(Xp, Wuxp, uxbf);
  }

  // Phase 2: sequential recurrence (2 graph nodes per step)
  for (int s = 0; s < SEQ; ++s) {
    lstm_gates_kernel<<<64, 256, 0, stream>>>(
        hbf, Whp, uxbf, b_ux, b_h, ctbuf, gates, zct, s);
    lstm_update_kernel<<<16, 256, 0, stream>>>(
        zct, Wump, uxbf, b_ux, b_um, gates, cbuf, ctbuf, hbf, out, s);
  }

  finish_kernel<<<(BATCH * HDIM + 255) / 256, 256, 0, stream>>>(cbuf, out);
}
